// GPT_3642132267471
// MI455X (gfx1250) — matrix operand order repaired and checked by reading, not compiled, not run on hardware
//
#include <hip/hip_runtime.h>
#include <math.h>

// ---------------- model constants ----------------
static constexpr int Lc = 8;
static constexpr int Hc = 12;
static constexpr int Dc = 768;
static constexpr int Tc = 2048;
static constexpr int Vc = 50257;
static constexpr int HD = 64;              // head dim

// ---------------- WMMA types ----------------
typedef __attribute__((ext_vector_type(16))) __bf16 v16bf;
typedef __attribute__((ext_vector_type(8)))  float  v8f;

// Native bf16 convert (RNE) -> backend uses v_cvt_*bf16_f32
__device__ inline unsigned short f2bf(float f) {
  union { __bf16 b; unsigned short u; } v;
  v.b = (__bf16)f;
  return v.u;
}
__device__ inline float bf2f(unsigned short h) {
  union { unsigned u; float f; } v; v.u = (unsigned)h << 16; return v.f;
}

union FragU { v16bf v; uint4 q[2]; };

// ---------------- async global->LDS staging (CDNA5) ----------------
// GLOBAL_LOAD_ASYNC_TO_LDS_B128: per-lane DMA into LDS, tracked with
// ASYNCcnt (must be drained before the barrier that publishes the tile).
// Builtin takes v4i pointers (per compiler diagnostic): AS1 src, AS3 dst.
// Pointer casts go through integers: generic global addrs are identity-
// mapped; the LDS address is the low 32 address bits, so truncation gives the LDS offset.
#if defined(__has_builtin)
#if __has_builtin(__builtin_amdgcn_global_load_async_to_lds_b128)
#define HAVE_ASYNC_LDS 1
#endif
#endif
#ifndef HAVE_ASYNC_LDS
#define HAVE_ASYNC_LDS 0
#endif

typedef int v4i_ls __attribute__((vector_size(16)));
typedef v4i_ls __attribute__((address_space(1))) g_v4i;   // global
typedef v4i_ls __attribute__((address_space(3))) l_v4i;   // LDS

__device__ inline void lds_copy_b128(unsigned short* dst,
                                     const unsigned short* src) {
#if HAVE_ASYNC_LDS
  g_v4i* gp = (g_v4i*)(unsigned long long)src;
  l_v4i* lp = (l_v4i*)(unsigned int)(unsigned long long)dst;
  __builtin_amdgcn_global_load_async_to_lds_b128(gp, lp, 0, 0);
#else
  *(uint4*)dst = *(const uint4*)src;
#endif
}
__device__ inline void lds_copy_fence() {
#if HAVE_ASYNC_LDS
#if __has_builtin(__builtin_amdgcn_s_wait_asynccnt)
  __builtin_amdgcn_s_wait_asynccnt(0);
#else
  asm volatile("s_wait_asynccnt 0x0" ::: "memory");
#endif
#endif
}

// 16-lane butterfly reduction via DPP16 (no LDS traffic).
template<int CTRL>
__device__ inline float dppf(float x) {
  union { float f; int i; } a; a.f = x;
  a.i = __builtin_amdgcn_mov_dpp(a.i, CTRL, 0xF, 0xF, true);
  return a.f;
}
__device__ inline float red16_max(float x) {
  x = fmaxf(x, dppf<0xB1>(x));    // quad_perm(1,0,3,2)
  x = fmaxf(x, dppf<0x4E>(x));    // quad_perm(2,3,0,1)
  x = fmaxf(x, dppf<0x141>(x));   // row_half_mirror
  x = fmaxf(x, dppf<0x140>(x));   // row_mirror
  return x;
}
__device__ inline float red16_sum(float x) {
  x += dppf<0xB1>(x);
  x += dppf<0x4E>(x);
  x += dppf<0x141>(x);
  x += dppf<0x140>(x);
  return x;
}

// =====================================================================
// Weight GEMM:  C[M,N] = epilogue( A[M,K] @ W^T ),  W f32 [N,K]
//   EPI 0: store bf16 | EPI 2: f32 + resid | EPI 3: exact-erf GELU->bf16
// Block tile 64x128, 8 waves, each wave 32x32 (2x2 WMMAs), K-step 32.
// =====================================================================
template<int EPI>
__global__ __launch_bounds__(256) void gemm_wmma(
    const unsigned short* __restrict__ A, int lda,
    const float* __restrict__ W, int ldw,
    void* __restrict__ Cv, int ldc,
    const float* __restrict__ resid,
    int M, int N, int K)
{
  constexpr int BM = 64, BN = 128, BK = 32, MT = 2, NT = 2;

  __shared__ unsigned short sA[BM][BK + 8];   // [m][k], 80B row stride
  __shared__ unsigned short sB[BN][BK + 8];   // [n][k]

  const int tid  = threadIdx.x;
  const int lane = tid & 31;
  const int wave = tid >> 5;
  const int wm   = wave >> 2;      // 0..1
  const int wn   = wave & 3;       // 0..3
  const int hi   = lane >> 4;
  const int l15  = lane & 15;
  const int m0   = blockIdx.y * BM;
  const int n0   = blockIdx.x * BN;

  v8f acc[MT][NT];
#pragma unroll
  for (int mt = 0; mt < MT; ++mt)
#pragma unroll
    for (int nt = 0; nt < NT; ++nt)
      acc[mt][nt] = (v8f){};

  for (int kk = 0; kk < K; kk += BK) {
    // ---- stage A tile: 64x32 bf16, async b128 per thread ----
    {
      const int row = tid >> 2;
      const int cg  = (tid & 3) * 8;
      lds_copy_b128(&sA[row][cg], &A[(size_t)(m0 + row) * lda + kk + cg]);
      if (kk + BK < K)
        __builtin_prefetch(&A[(size_t)(m0 + row) * lda + kk + BK + cg], 0, 1);
    }
    // ---- stage W tile (f32 -> bf16, native cvt) as sB[n][k] ----
    {
      const int row = tid >> 3;
      const int cg  = (tid & 7) * 4;
#pragma unroll
      for (int p = 0; p < BN / 32; ++p) {
        const int n = row + p * 32;
        const float4 w4 = *(const float4*)&W[(size_t)(n0 + n) * ldw + kk + cg];
        union { uint2 u2; unsigned short h[4]; } pk;
        pk.h[0] = f2bf(w4.x); pk.h[1] = f2bf(w4.y);
        pk.h[2] = f2bf(w4.z); pk.h[3] = f2bf(w4.w);
        *(uint2*)&sB[n][cg] = pk.u2;
        if (kk + BK < K)
          __builtin_prefetch(&W[(size_t)(n0 + n) * ldw + kk + BK + cg], 0, 1);
      }
    }
    lds_copy_fence();
    __syncthreads();

    FragU fa[MT], fb[NT];
#pragma unroll
    for (int mt = 0; mt < MT; ++mt) {
      const int mrow = (wm * MT + mt) * 16 + l15;
      fa[mt].q[0] = *(const uint4*)&sA[mrow][hi * 8];
      fa[mt].q[1] = *(const uint4*)&sA[mrow][16 + hi * 8];
    }
#pragma unroll
    for (int nt = 0; nt < NT; ++nt) {
      const int ncol = (wn * NT + nt) * 16 + l15;
      fb[nt].q[0] = *(const uint4*)&sB[ncol][hi * 8];
      fb[nt].q[1] = *(const uint4*)&sB[ncol][16 + hi * 8];
    }

#pragma unroll
    for (int mt = 0; mt < MT; ++mt)
#pragma unroll
      for (int nt = 0; nt < NT; ++nt)
        acc[mt][nt] = __builtin_amdgcn_wmma_f32_16x16x32_bf16(
            false, fa[mt].v, false, fb[nt].v, (short)0, acc[mt][nt],
            false, false);

    __syncthreads();
  }

  // ---- epilogue: C layout M = (lane>>4)*8 + r, N = lane&15 ----
#pragma unroll
  for (int mt = 0; mt < MT; ++mt) {
#pragma unroll
    for (int nt = 0; nt < NT; ++nt) {
#pragma unroll
      for (int r = 0; r < 8; ++r) {
        const int m = m0 + (wm * MT + mt) * 16 + hi * 8 + r;
        const int n = n0 + (wn * NT + nt) * 16 + l15;
        const float v = acc[mt][nt][r];
        if (EPI == 0) {
          ((unsigned short*)Cv)[(size_t)m * ldc + n] = f2bf(v);
        } else if (EPI == 2) {
          ((float*)Cv)[(size_t)m * ldc + n] = resid[(size_t)m * ldc + n] + v;
        } else {
          const float g = 0.5f * v * (1.0f + erff(v * 0.70710678118654752f));
          ((unsigned short*)Cv)[(size_t)m * ldc + n] = f2bf(g);
        }
      }
    }
  }
}

// =====================================================================
// Fused flash attention (causal), one head per blockIdx.y.
// 128 queries per block, 8 waves; wave w owns queries [q0+16w, q0+16w+16)
// and the full 64-dim output (4 f32 WMMA accumulators).
// =====================================================================
__global__ __launch_bounds__(256) void fattn_k(
    const unsigned short* __restrict__ qkv,  // [T, 3D] bf16
    unsigned short* __restrict__ y,          // [T, D]  bf16
    int T)
{
  constexpr int BQ = 128;
  constexpr int LDQ = 3 * Dc;
  const int h   = blockIdx.y;
  const int q0  = blockIdx.x * BQ;
  const int tid = threadIdx.x;
  const int lane = tid & 31;
  const int w    = tid >> 5;
  const int hi   = lane >> 4;
  const int l15  = lane & 15;

  __shared__ unsigned short sQ[BQ][HD + 8];   // [q][d]   144B stride
  __shared__ unsigned short sK[32][HD + 8];   // [key][d]
  __shared__ unsigned short sVt[HD][32 + 8];  // [d][key] 80B stride
  __shared__ unsigned short sP[8][16][32 + 8];// per-wave P tile [q][key]

  // ---- stage Q once, pre-scaled by 0.125 (exact: exponent -3) ----
  {
    const int row = tid >> 3;        // 0..31
    const int cg  = (tid & 7) * 8;   // 0..56
#pragma unroll
    for (int p = 0; p < 4; ++p) {
      const int r = row + p * 32;
      union { uint4 q; unsigned short hh[8]; } u;
      u.q = *(const uint4*)&qkv[(size_t)(q0 + r) * LDQ + h * HD + cg];
#pragma unroll
      for (int j = 0; j < 8; ++j) u.hh[j] = f2bf(bf2f(u.hh[j]) * 0.125f);
      *(uint4*)&sQ[r][cg] = u.q;
    }
  }

  float m_run[8], l_run[8];
#pragma unroll
  for (int r = 0; r < 8; ++r) { m_run[r] = -1e30f; l_run[r] = 0.0f; }
  v8f accO[4];
#pragma unroll
  for (int nt = 0; nt < 4; ++nt) accO[nt] = (v8f){};

  const int qw  = q0 + w * 16;         // wave's first query
  const int nkb = (q0 + BQ) / 32;      // causal: keys < q0+128

  __syncthreads();

  for (int kb = 0; kb < nkb; ++kb) {
    const int k0 = kb * 32;
    // ---- stage K block (async DMA) and transposed V block ----
    {
      const int krow = tid >> 3;       // 0..31
      const int cg   = (tid & 7) * 8;  // 0..56
      lds_copy_b128(&sK[krow][cg],
                    &qkv[(size_t)(k0 + krow) * LDQ + Dc + h * HD + cg]);
      union { uint4 q; unsigned short hh[8]; } uv;
      uv.q = *(const uint4*)&qkv[(size_t)(k0 + krow) * LDQ + 2 * Dc + h * HD + cg];
#pragma unroll
      for (int j = 0; j < 8; ++j) sVt[cg + j][krow] = uv.hh[j];
    }
    lds_copy_fence();
    __syncthreads();

    if (k0 <= qw + 15) {   // wave-uniform causal skip
      // ---- scores: S[16q x 32k] = Q16x64 @ K^T ----
      FragU fq[2], fk;
      const int qrow = w * 16 + l15;
      v8f accS[2];
#pragma unroll
      for (int kt = 0; kt < 2; ++kt) accS[kt] = (v8f){};
      fq[0].q[0] = *(const uint4*)&sQ[qrow][hi * 8];
      fq[0].q[1] = *(const uint4*)&sQ[qrow][16 + hi * 8];
      fq[1].q[0] = *(const uint4*)&sQ[qrow][32 + hi * 8];
      fq[1].q[1] = *(const uint4*)&sQ[qrow][48 + hi * 8];
#pragma unroll
      for (int kt = 0; kt < 2; ++kt) {
        const int krow_f = kt * 16 + l15;
#pragma unroll
        for (int d32 = 0; d32 < 2; ++d32) {
          fk.q[0] = *(const uint4*)&sK[krow_f][d32 * 32 + hi * 8];
          fk.q[1] = *(const uint4*)&sK[krow_f][d32 * 32 + 16 + hi * 8];
          accS[kt] = __builtin_amdgcn_wmma_f32_16x16x32_bf16(
              false, fq[d32].v, false, fk.v, (short)0, accS[kt], false, false);
        }
      }

      // ---- online softmax, in-register (DPP16 reductions) ----
      float p0[8], p1[8], alpha[8];
#pragma unroll
      for (int r = 0; r < 8; ++r) {
        const int qg = qw + hi * 8 + r;
        float s0 = accS[0][r]; if (k0 + l15 > qg)      s0 = -1e30f;
        float s1 = accS[1][r]; if (k0 + 16 + l15 > qg) s1 = -1e30f;
        const float mx = red16_max(fmaxf(s0, s1));
        const float mnew = fmaxf(m_run[r], mx);
        alpha[r] = __expf(m_run[r] - mnew);
        p0[r] = __expf(s0 - mnew);
        p1[r] = __expf(s1 - mnew);
        const float ps = red16_sum(p0[r] + p1[r]);
        l_run[r] = l_run[r] * alpha[r] + ps;
        m_run[r] = mnew;
      }

      // ---- P tile -> private LDS slice (C-layout -> A-layout bounce) ----
#pragma unroll
      for (int r = 0; r < 8; ++r) {
        sP[w][hi * 8 + r][l15]      = f2bf(p0[r]);
        sP[w][hi * 8 + r][16 + l15] = f2bf(p1[r]);
      }
      // rescale O accumulators
#pragma unroll
      for (int nt = 0; nt < 4; ++nt)
#pragma unroll
        for (int r = 0; r < 8; ++r) accO[nt][r] *= alpha[r];

      // ---- O += P @ V ----
      FragU fp, fv;
      fp.q[0] = *(const uint4*)&sP[w][l15][hi * 8];
      fp.q[1] = *(const uint4*)&sP[w][l15][16 + hi * 8];
#pragma unroll
      for (int nt = 0; nt < 4; ++nt) {
        fv.q[0] = *(const uint4*)&sVt[nt * 16 + l15][hi * 8];
        fv.q[1] = *(const uint4*)&sVt[nt * 16 + l15][16 + hi * 8];
        accO[nt] = __builtin_amdgcn_wmma_f32_16x16x32_bf16(
            false, fp.v, false, fv.v, (short)0, accO[nt], false, false);
      }
    }
    __syncthreads();
  }

  // ---- normalize and store y[q, h*64 + d] ----
  float inv[8];
#pragma unroll
  for (int r = 0; r < 8; ++r) inv[r] = 1.0f / l_run[r];
#pragma unroll
  for (int nt = 0; nt < 4; ++nt) {
#pragma unroll
    for (int r = 0; r < 8; ++r) {
      const int qg = qw + hi * 8 + r;
      const int d  = nt * 16 + l15;
      y[(size_t)qg * Dc + h * HD + d] = f2bf(accO[nt][r] * inv[r]);
    }
  }
}

// =====================================================================
// Embedding: x[t,d] = wte[idx[t],d] + wpe[t,d]
// =====================================================================
__global__ void embed_k(const int* __restrict__ idx, const float* __restrict__ wte,
                        const float* __restrict__ wpe, float* __restrict__ x,
                        int T, int D) {
  const int t = blockIdx.x;
  const int d = blockIdx.y * blockDim.x + threadIdx.x;
  if (d < D)
    x[(size_t)t * D + d] = wte[(size_t)idx[t] * D + d] + wpe[(size_t)t * D + d];
}

// =====================================================================
// LayerNorm (two-pass, matches reference): one block per row
// =====================================================================
__global__ __launch_bounds__(256) void ln_rows(
    const float* __restrict__ x, const float* __restrict__ w,
    const float* __restrict__ b, unsigned short* __restrict__ obf,
    float* __restrict__ of32, int D) {
  const int row = blockIdx.x;
  const int tid = threadIdx.x;
  const float* xp = x + (size_t)row * D;
  __shared__ float red[256];

  float s = 0.0f;
  for (int i = tid; i < D; i += 256) s += xp[i];
  red[tid] = s; __syncthreads();
  for (int off = 128; off > 0; off >>= 1) {
    if (tid < off) red[tid] += red[tid + off];
    __syncthreads();
  }
  const float mu = red[0] / (float)D;
  __syncthreads();

  float s2 = 0.0f;
  for (int i = tid; i < D; i += 256) { const float d = xp[i] - mu; s2 += d * d; }
  red[tid] = s2; __syncthreads();
  for (int off = 128; off > 0; off >>= 1) {
    if (tid < off) red[tid] += red[tid + off];
    __syncthreads();
  }
  const float rs = rsqrtf(red[0] / (float)D + 1e-5f);

  for (int i = tid; i < D; i += 256) {
    const float y = (xp[i] - mu) * rs * w[i] + b[i];
    obf[(size_t)row * D + i] = f2bf(y);
    if (of32) of32[(size_t)row * D + i] = y;
  }
}

// =====================================================================
// Tied lm_head GEMV (last token only): one wave per vocab row, f32,
// float4 (b128) streaming loads: 768 = 32 lanes * 6 * float4 exactly.
// =====================================================================
__global__ __launch_bounds__(256) void lmhead_k(
    const float* __restrict__ wte, const float* __restrict__ xf,
    float* __restrict__ out, int V, int D) {
  const int wave = threadIdx.x >> 5;
  const int lane = threadIdx.x & 31;
  const int v = blockIdx.x * 8 + wave;
  if (v >= V) return;
  const float4* wr4 = (const float4*)(wte + (size_t)v * D);
  const float4* x4  = (const float4*)xf;
  float s = 0.0f;
#pragma unroll
  for (int i = 0; i < 6; ++i) {
    const float4 a = wr4[lane + i * 32];
    const float4 b = x4[lane + i * 32];
    s += a.x * b.x + a.y * b.y + a.z * b.z + a.w * b.w;
  }
  for (int off = 16; off > 0; off >>= 1) s += __shfl_down(s, off, 32);
  if (lane == 0) out[v] = s;
}

// =====================================================================
// Orchestration
// =====================================================================
extern "C" void kernel_launch(void* const* d_in, const int* in_sizes, int n_in,
                              void* d_out, int out_size, void* d_ws, size_t ws_size,
                              hipStream_t stream) {
  (void)in_sizes; (void)n_in; (void)out_size; (void)ws_size;
  const int*   idx          = (const int*)d_in[0];
  const float* wte          = (const float*)d_in[1];
  const float* wpe          = (const float*)d_in[2];
  const float* ln1_w        = (const float*)d_in[3];
  const float* ln1_b        = (const float*)d_in[4];
  const float* attn_w       = (const float*)d_in[5];
  const float* attn_proj_w  = (const float*)d_in[6];
  const float* ln2_w        = (const float*)d_in[7];
  const float* ln2_b        = (const float*)d_in[8];
  const float* fc_w         = (const float*)d_in[9];
  const float* mlp_proj_w   = (const float*)d_in[10];
  const float* lnf_w        = (const float*)d_in[11];
  const float* lnf_b        = (const float*)d_in[12];

  // ---- workspace layout (bytes) ----
  char* ws = (char*)d_ws;
  constexpr size_t OFF_X    = 0;                                  // f32  [T,D]
  constexpr size_t OFF_HLN  = OFF_X    + (size_t)Tc * Dc * 4;     // bf16 [T,D]
  constexpr size_t OFF_QKV  = OFF_HLN  + (size_t)Tc * Dc * 2;     // bf16 [T,3D]
  constexpr size_t OFF_Y    = OFF_QKV  + (size_t)Tc * 3 * Dc * 2; // bf16 [T,D]
  constexpr size_t OFF_MACT = OFF_Y    + (size_t)Tc * Dc * 2;     // bf16 [T,4D]
  constexpr size_t OFF_XF   = OFF_MACT + (size_t)Tc * 4 * Dc * 2; // f32  [D]

  float*          x    = (float*)(ws + OFF_X);
  unsigned short* hln  = (unsigned short*)(ws + OFF_HLN);
  unsigned short* qkv  = (unsigned short*)(ws + OFF_QKV);
  unsigned short* ybuf = (unsigned short*)(ws + OFF_Y);
  unsigned short* mact = (unsigned short*)(ws + OFF_MACT);
  float*          xf   = (float*)(ws + OFF_XF);

  // ---- embeddings ----
  embed_k<<<dim3(Tc, Dc / 256), 256, 0, stream>>>(idx, wte, wpe, x, Tc, Dc);

  for (int l = 0; l < Lc; ++l) {
    const float* aw  = attn_w      + (size_t)l * 3 * Dc * Dc;
    const float* apw = attn_proj_w + (size_t)l * Dc * Dc;
    const float* fw  = fc_w        + (size_t)l * 4 * Dc * Dc;
    const float* pw  = mlp_proj_w  + (size_t)l * Dc * 4 * Dc;

    // ln1 -> hln (bf16)
    ln_rows<<<Tc, 256, 0, stream>>>(x, ln1_w + l * Dc, ln1_b + l * Dc,
                                    hln, nullptr, Dc);
    // qkv = hln @ aw^T  -> bf16 [T,3D]
    gemm_wmma<0><<<dim3(3 * Dc / 128, Tc / 64), 256, 0, stream>>>(
        hln, Dc, aw, Dc, qkv, 3 * Dc, nullptr, Tc, 3 * Dc, Dc);

    // fused causal flash attention over all heads -> ybuf bf16 [T,D]
    fattn_k<<<dim3(Tc / 128, Hc), 256, 0, stream>>>(qkv, ybuf, Tc);

    // x += y @ apw^T
    gemm_wmma<2><<<dim3(Dc / 128, Tc / 64), 256, 0, stream>>>(
        ybuf, Dc, apw, Dc, x, Dc, x, Tc, Dc, Dc);

    // ln2 -> hln
    ln_rows<<<Tc, 256, 0, stream>>>(x, ln2_w + l * Dc, ln2_b + l * Dc,
                                    hln, nullptr, Dc);
    // mact = gelu(hln @ fw^T) -> bf16 [T,4D]
    gemm_wmma<3><<<dim3(4 * Dc / 128, Tc / 64), 256, 0, stream>>>(
        hln, Dc, fw, Dc, mact, 4 * Dc, nullptr, Tc, 4 * Dc, Dc);
    // x += mact @ pw^T
    gemm_wmma<2><<<dim3(Dc / 128, Tc / 64), 256, 0, stream>>>(
        mact, 4 * Dc, pw, 4 * Dc, x, Dc, x, Tc, Dc, 4 * Dc);
  }

  // final LN of last token -> xf (f32)
  ln_rows<<<1, 256, 0, stream>>>(x + (size_t)(Tc - 1) * Dc, lnf_w, lnf_b,
                                 hln, xf, Dc);
  // logits = xf @ wte^T  -> d_out f32 [V]
  lmhead_k<<<(Vc + 7) / 8, 256, 0, stream>>>(wte, xf, (float*)d_out, Vc, Dc);
}
